// MLinear_87608742903883
// MI455X (gfx1250) — matrix operand order repaired and checked by reading, not compiled, not run on hardware
//
#include <hip/hip_runtime.h>

// CDNA5 / gfx1250: wave32, WMMA 16x16x32 bf16 with f32 accumulate,
// async global->LDS DMA (ASYNCcnt) for the weight tiles.
typedef __attribute__((ext_vector_type(16))) __bf16 v16bf;
typedef __attribute__((ext_vector_type(8)))  __bf16 v8bf;
typedef __attribute__((ext_vector_type(4)))  __bf16 v4bf;
typedef __attribute__((ext_vector_type(8)))  float  v8f;

#define TM 128
#define TN 128
#define TK 32
#define LSTR 40   // ushorts per LDS row: 32 data + 8 pad = 80B (16B-aligned rows)

// ---------------------------------------------------------------------------
// Prologue: pack the three Matryoshka weight blocks into transposed bf16
// hi/lo planes in workspace:  WT_hi[n][k], WT_lo[n][k]  (n-major, k stride 2048).
// ---------------------------------------------------------------------------
__global__ __launch_bounds__(256)
void pack_weights(const float* __restrict__ w0,
                  const float* __restrict__ w1,
                  const float* __restrict__ w2,
                  unsigned short* __restrict__ wtH,
                  unsigned short* __restrict__ wtL)
{
    const int n  = blockIdx.x;          // 0..2047 output column
    const int k0 = threadIdx.x * 8;     // 8 consecutive k per thread

    int Keff, wStride, wCol;
    const float* W;
    if (n < 512)       { Keff = 512;  W = w0; wStride = 512;  wCol = 0;    }
    else if (n < 1024) { Keff = 1024; W = w1; wStride = 512;  wCol = 512;  }
    else               { Keff = 2048; W = w2; wStride = 1024; wCol = 1024; }
    if (k0 >= Keff) return;             // zero-region never read by GEMM

    v8bf hv, lv;
    #pragma unroll
    for (int i = 0; i < 8; ++i) {
        float f = W[(size_t)(k0 + i) * wStride + (n - wCol)];
        __bf16 h = (__bf16)f;           // RNE
        hv[i] = h;
        lv[i] = (__bf16)(f - (float)h); // residual
    }
    *(v8bf*)(wtH + (size_t)n * 2048 + k0) = hv;
    *(v8bf*)(wtL + (size_t)n * 2048 + k0) = lv;
}

// ---------------------------------------------------------------------------
// Main GEMM: C[16384,2048] = X * W_matryoshka + bias, bf16 3-term split,
// 128x128 tile / 8 waves (4x2), double-buffered LDS, async DMA for weights.
// ---------------------------------------------------------------------------
__global__ __launch_bounds__(256)
void mlinear_wmma_bf16x3(const float* __restrict__ x,
                         const unsigned short* __restrict__ wtH,
                         const unsigned short* __restrict__ wtL,
                         const float* __restrict__ bias,
                         float* __restrict__ out)
{
    __shared__ unsigned short sA[2][2][TM * LSTR];   // [buf][hi/lo]
    __shared__ unsigned short sB[2][2][TN * LSTR];   // [buf][hi/lo], [n][k] layout

    const int tid   = threadIdx.x;
    const int nBase = blockIdx.x * TN;
    const int mBase = blockIdx.y * TM;

    int Keff;
    if (nBase < 512)       Keff = 512;
    else if (nBase < 1024) Keff = 1024;
    else                   Keff = 2048;

    const int lane = tid & 31;
    const int wave = tid >> 5;
    const int wm   = (wave & 3) * 32;   // wave row offset in tile
    const int wn   = (wave >> 2) * 64;  // wave col offset in tile
    const int lh   = lane >> 4;
    const int l16  = lane & 15;

    v8f acc[2][4];
    #pragma unroll
    for (int i = 0; i < 2; ++i)
        #pragma unroll
        for (int j = 0; j < 4; ++j)
            #pragma unroll
            for (int r = 0; r < 8; ++r)
                acc[i][j][r] = 0.0f;

    // -------- per-thread async-DMA descriptors for the B tile (invariant) ----
    // B tile = 128 n-rows x 32 k of bf16, hi+lo planes: 1024 16B chunks / 256 thr.
    unsigned voffB[4];   // byte offset into plane (k0 folded into scalar base)
    unsigned ldsB[4];    // LDS byte offset of chunk in buffer 0
    bool     plB[4];     // which plane
    #pragma unroll
    for (int it = 0; it < 4; ++it) {
        int idx   = tid + it * 256;
        int plane = idx >> 9;              // 0=hi, 1=lo
        int r     = (idx >> 2) & 127;      // n within tile
        int c     = idx & 3;               // 8-element k chunk
        voffB[it] = ((unsigned)(nBase + r) << 12) + (unsigned)c * 16u; // row = 4096B
        ldsB[it]  = (unsigned)(unsigned long long)(void*)
                        &sB[0][plane][r * LSTR + c * 8];  // addr[31:0] = LDS offset
        plB[it]   = plane != 0;
    }
    const unsigned BBUF = (unsigned)sizeof(sB[0]);        // buf0 -> buf1 delta

    auto stage_B_async = [&](int buf, int k0) {
        const unsigned short* baseH = wtH + k0;           // uniform k advance in SGPR
        const unsigned short* baseL = wtL + k0;
        const unsigned add = buf ? BBUF : 0u;
        #pragma unroll
        for (int it = 0; it < 4; ++it) {
            const unsigned short* sbase = plB[it] ? baseL : baseH;
            asm volatile("global_load_async_to_lds_b128 %0, %1, %2 offset:0"
                         :: "v"(ldsB[it] + add), "v"(voffB[it]), "s"(sbase)
                         : "memory");
        }
    };

    // -------- X staging: global f32 -> regs -> bf16 hi/lo in LDS ------------
    float4 ra[4];
    auto load_A = [&](int k0) {
        #pragma unroll
        for (int it = 0; it < 4; ++it) {
            int idx = tid + it * 256;
            int row = idx >> 3, kq = (idx & 7) << 2;
            ra[it] = *(const float4*)(x + (size_t)(mBase + row) * 2048 + (k0 + kq));
        }
    };
    auto store_A = [&](int buf) {
        #pragma unroll
        for (int it = 0; it < 4; ++it) {
            int idx = tid + it * 256;
            int row = idx >> 3, kq = (idx & 7) << 2;
            float4 v = ra[it];
            v4bf h, l;
            h.x = (__bf16)v.x; h.y = (__bf16)v.y; h.z = (__bf16)v.z; h.w = (__bf16)v.w;
            l.x = (__bf16)(v.x - (float)h.x);
            l.y = (__bf16)(v.y - (float)h.y);
            l.z = (__bf16)(v.z - (float)h.z);
            l.w = (__bf16)(v.w - (float)h.w);
            *(v4bf*)&sA[buf][0][row * LSTR + kq] = h;
            *(v4bf*)&sA[buf][1][row * LSTR + kq] = l;
        }
    };

    auto compute = [&](int buf) {
        // B frags first (reused across both i groups)
        v16bf bH[4], bL[4];
        #pragma unroll
        for (int j = 0; j < 4; ++j) {
            const unsigned short* p = &sB[buf][0][(wn + j * 16 + l16) * LSTR + lh * 8];
            ((uint4*)&bH[j])[0] = *(const uint4*)(p);
            ((uint4*)&bH[j])[1] = *(const uint4*)(p + 16);
            const unsigned short* q = &sB[buf][1][(wn + j * 16 + l16) * LSTR + lh * 8];
            ((uint4*)&bL[j])[0] = *(const uint4*)(q);
            ((uint4*)&bL[j])[1] = *(const uint4*)(q + 16);
        }
        #pragma unroll
        for (int i = 0; i < 2; ++i) {
            v16bf aH, aL;   // short live range: loaded per i group
            const unsigned short* p = &sA[buf][0][(wm + i * 16 + l16) * LSTR + lh * 8];
            ((uint4*)&aH)[0] = *(const uint4*)(p);
            ((uint4*)&aH)[1] = *(const uint4*)(p + 16);
            const unsigned short* q = &sA[buf][1][(wm + i * 16 + l16) * LSTR + lh * 8];
            ((uint4*)&aL)[0] = *(const uint4*)(q);
            ((uint4*)&aL)[1] = *(const uint4*)(q + 16);
            #pragma unroll
            for (int j = 0; j < 4; ++j) {
                acc[i][j] = __builtin_amdgcn_wmma_f32_16x16x32_bf16(
                    false, aH, false, bH[j], (short)0, acc[i][j], false, false);
                acc[i][j] = __builtin_amdgcn_wmma_f32_16x16x32_bf16(
                    false, aH, false, bL[j], (short)0, acc[i][j], false, false);
                acc[i][j] = __builtin_amdgcn_wmma_f32_16x16x32_bf16(
                    false, aL, false, bH[j], (short)0, acc[i][j], false, false);
            }
        }
    };

    // -------- software pipeline: async B DMA + reg-prefetched A -------------
    stage_B_async(0, 0);
    load_A(0);
    store_A(0);
    asm volatile("s_wait_asynccnt 0x0" ::: "memory");
    __syncthreads();

    int buf = 0;
    #pragma clang loop unroll(disable)
    for (int k0 = TK; k0 < Keff; k0 += TK) {
        stage_B_async(buf ^ 1, k0);   // DMA next weight tile (overlaps compute)
        load_A(k0);                   // issue X loads (hidden under WMMA)
        compute(buf);                 // 24 x v_wmma
        store_A(buf ^ 1);             // convert + stage next X tile
        asm volatile("s_wait_asynccnt 0x0" ::: "memory");
        __syncthreads();
        buf ^= 1;
    }
    compute(buf);

    // ---- epilogue: bias add + streaming (non-temporal) stores ----
    #pragma unroll
    for (int j = 0; j < 4; ++j) {
        const int n  = nBase + wn + j * 16 + l16;
        const float bv = bias[n];
        #pragma unroll
        for (int i = 0; i < 2; ++i) {
            const int m0 = mBase + wm + i * 16 + lh * 8;
            #pragma unroll
            for (int r = 0; r < 8; ++r)
                __builtin_nontemporal_store(acc[i][j][r] + bv,
                                            out + (size_t)(m0 + r) * 2048 + n);
        }
    }
}

extern "C" void kernel_launch(void* const* d_in, const int* in_sizes, int n_in,
                              void* d_out, int out_size, void* d_ws, size_t ws_size,
                              hipStream_t stream) {
    const float* x    = (const float*)d_in[0];  // (4,4096,2048)
    const float* w0   = (const float*)d_in[1];  // (512,512)
    const float* w1   = (const float*)d_in[2];  // (1024,512)
    const float* w2   = (const float*)d_in[3];  // (2048,1024)
    const float* bias = (const float*)d_in[4];  // (2048,)
    float* out = (float*)d_out;

    // workspace: two transposed bf16 planes, 2048*2048 ushorts each (16.8 MB)
    unsigned short* wtH = (unsigned short*)d_ws;
    unsigned short* wtL = wtH + (size_t)2048 * 2048;

    pack_weights<<<dim3(2048), 256, 0, stream>>>(w0, w1, w2, wtH, wtL);

    const int M = in_sizes[0] / 2048;           // 16384 rows
    dim3 grid(2048 / TN, M / TM);               // (16, 128); N fastest for L2 reuse of X
    mlinear_wmma_bf16x3<<<grid, 256, 0, stream>>>(x, wtH, wtL, bias, out);
}
